// GraphormerMultiheadAttention_36498632081516
// MI455X (gfx1250) — matrix operand order repaired and checked by reading, not compiled, not run on hardware
//
#include <hip/hip_runtime.h>

typedef __attribute__((ext_vector_type(16))) _Float16 v16h;
typedef __attribute__((ext_vector_type(8)))  _Float16 h8;
typedef __attribute__((ext_vector_type(4)))  _Float16 h4;
typedef __attribute__((ext_vector_type(8)))  float    v8f;
typedef __attribute__((ext_vector_type(4)))  float    f4;
typedef __attribute__((ext_vector_type(4)))  int      vi4;

#define NH    12
#define HD    64
#define NSEQ  512
#define BATCH 16
#define HDIM  768

#if defined(__gfx1250__) && __has_builtin(__builtin_amdgcn_global_load_async_to_lds_b128) && __has_builtin(__builtin_amdgcn_s_wait_asynccnt)
#define USE_ASYNC_COPY 1
#endif

#ifdef USE_ASYNC_COPY
// signature (from hipcc diagnostic): param0 = v4i* in global AS, param1 = LDS side,
// then two integer immediates (offset, cpol).
static __device__ __forceinline__ void async_copy_b128(const void* g, void* l) {
    __builtin_amdgcn_global_load_async_to_lds_b128(
        (__attribute__((address_space(1))) vi4*)g,
        (__attribute__((address_space(3))) vi4*)l,
        0, 0);
}
#endif

static __device__ __forceinline__ v8f vzero8() {
    v8f z;
#pragma unroll
    for (int i = 0; i < 8; ++i) z[i] = 0.0f;
    return z;
}

static __device__ __forceinline__ v16h mkv16(h8 a, h8 b) {
    return __builtin_shufflevector(a, b, 0,1,2,3,4,5,6,7,8,9,10,11,12,13,14,15);
}

static __device__ __forceinline__ v8f wmma_f16(v16h a, v16h b, v8f c) {
    // D = A(16x32 f16) x B(32x16 f16) + C(16x16 f32)
    return __builtin_amdgcn_wmma_f32_16x16x32_f16(false, a, false, b, (short)0, c, false, false);
}

// ---------------------------------------------------------------------------
// Kernel 1: fused QKV projection.  Y = X @ W{q,k,v} + b{q,k,v}
// X: (8192 x 768) f32 row-major (row = n*16 + b).  Output: f16 (B,NH,N,HD).
// Block tile M=64, N=128, K-step 32; 8 waves (2x4), each 32x32 (2x2 WMMA).
// W tile staged TRANSPOSED in LDS so B-fragments are contiguous along k.
// ---------------------------------------------------------------------------
__global__ __launch_bounds__(256) void qkv_kernel(
    const float* __restrict__ X,
    const float* __restrict__ Wq, const float* __restrict__ Wk, const float* __restrict__ Wv,
    const float* __restrict__ bq, const float* __restrict__ bk, const float* __restrict__ bv,
    _Float16* __restrict__ Qo, _Float16* __restrict__ Ko, _Float16* __restrict__ Vo)
{
    __shared__ alignas(16) _Float16 ldsX[64][40];     // rows x k (stride 80B)
    __shared__ alignas(16) _Float16 ldsWT[128][40];   // cols x k (transposed)

    const int mb  = blockIdx.x;          // 0..127
    const int nb  = blockIdx.y;          // 0..17
    const int sel = nb / 6;              // 0=Q 1=K 2=V
    const int ncc = (nb % 6) * 128;

    const float* W    = (sel == 0) ? Wq : (sel == 1) ? Wk : Wv;
    const float* bias = (sel == 0) ? bq : (sel == 1) ? bk : bv;
    _Float16*    dst  = (sel == 0) ? Qo : (sel == 1) ? Ko : Vo;

    const int tid   = threadIdx.x;
    const int wid   = tid >> 5;
    const int lane  = tid & 31;
    const int lhalf = lane & 15;
    const bool hi   = lane >= 16;
    const int wm    = wid & 1;
    const int wn    = wid >> 1;
    const int rowBase = mb * 64;

    // per-thread staging assignments
    const int wcol = tid & 127;          // 0..127 (column of W tile)
    const int wkq  = (tid >> 7) * 16;    // 0 or 16 (k range)

    v8f acc[2][2];
#pragma unroll
    for (int i = 0; i < 2; ++i)
#pragma unroll
        for (int j = 0; j < 2; ++j) acc[i][j] = vzero8();

    for (int k0 = 0; k0 < HDIM; k0 += 32) {
        __syncthreads();
        // X tile 64x32 f32 -> f16, packed h4 stores
#pragma unroll
        for (int it = 0; it < 2; ++it) {
            int i   = tid + it * 256;
            int row = i >> 3;
            int c4  = (i & 7) * 4;
            f4 v = *(const f4*)(X + (size_t)(rowBase + row) * HDIM + k0 + c4);
            h4 p;
#pragma unroll
            for (int j = 0; j < 4; ++j) p[j] = (_Float16)v[j];
            *(h4*)&ldsX[row][c4] = p;
        }
        // W tile 32x128 f32 -> f16, stored transposed (coalesced global reads)
        {
            h8 lo, hi8;
#pragma unroll
            for (int j = 0; j < 8; ++j)
                lo[j]  = (_Float16)W[(size_t)(k0 + wkq + j) * HDIM + ncc + wcol];
#pragma unroll
            for (int j = 0; j < 8; ++j)
                hi8[j] = (_Float16)W[(size_t)(k0 + wkq + 8 + j) * HDIM + ncc + wcol];
            *(h8*)&ldsWT[wcol][wkq]     = lo;
            *(h8*)&ldsWT[wcol][wkq + 8] = hi8;
        }
        __syncthreads();

        v16h a[2];
#pragma unroll
        for (int fm = 0; fm < 2; ++fm) {
            int row = wm * 32 + fm * 16 + lhalf;
            int db  = hi ? 8 : 0;
            a[fm] = mkv16(*(const h8*)&ldsX[row][db], *(const h8*)&ldsX[row][db + 16]);
        }
        v16h bf[2];
#pragma unroll
        for (int fn = 0; fn < 2; ++fn) {
            int col = wn * 32 + fn * 16 + lhalf;
            int kb2 = hi ? 8 : 0;
            bf[fn] = mkv16(*(const h8*)&ldsWT[col][kb2], *(const h8*)&ldsWT[col][kb2 + 16]);
        }
#pragma unroll
        for (int fm = 0; fm < 2; ++fm)
#pragma unroll
            for (int fn = 0; fn < 2; ++fn)
                acc[fm][fn] = wmma_f16(a[fm], bf[fn], acc[fm][fn]);
    }

    // Store as f16 into (B, NH, N, HD)
#pragma unroll
    for (int fm = 0; fm < 2; ++fm) {
#pragma unroll
        for (int fn = 0; fn < 2; ++fn) {
            int col = ncc + wn * 32 + fn * 16 + lhalf;   // within 768
            float bb = bias[col];
            int hh = col >> 6;
            int d  = col & 63;
#pragma unroll
            for (int r = 0; r < 8; ++r) {
                int rowI = rowBase + wm * 32 + fm * 16 + r + (hi ? 8 : 0);
                int n  = rowI >> 4;       // row = n*16 + b
                int bI = rowI & 15;
                float val = acc[fm][fn][r] + bb;
                dst[(((size_t)bI * NH + hh) * NSEQ + n) * HD + d] = (_Float16)val;
            }
        }
    }
}

// ---------------------------------------------------------------------------
// Kernel 2: flash attention per (b, h, 128-query block).
// 8 waves x 16 query rows; Q in registers.  K staged row-major (async copy
// on the gfx1250 ASYNCcnt path when available), V staged transposed so P.V
// B-fragments are contiguous.  Online softmax fused with bias + mask; the
// 201 MB bias stream is read exactly once and prefetched one block ahead.
// ---------------------------------------------------------------------------
__global__ __launch_bounds__(256) void attn_kernel(
    const _Float16* __restrict__ Q,   // (B,NH,N,HD) f16
    const _Float16* __restrict__ K,
    const _Float16* __restrict__ V,
    const float* __restrict__ bias,   // (B,NH,N,N) f32
    const unsigned char* __restrict__ mask,  // (B,N) bool
    _Float16* __restrict__ ctx)       // (N*B, HDIM) f16
{
    __shared__ alignas(16) _Float16 ldsK[64][72];        // keys x d
    __shared__ alignas(16) _Float16 ldsVT[64][72];       // d x keys (transposed)
    __shared__ alignas(16) _Float16 ldsP[8][16][72];     // per-wave P staging

    const int bid = blockIdx.x;
    const int h  = bid % NH;
    const int b  = (bid / NH) % BATCH;
    const int qb = bid / (NH * BATCH);       // 0..3

    const int tid   = threadIdx.x;
    const int wid   = tid >> 5;
    const int lane  = tid & 31;
    const int lhalf = lane & 15;
    const bool hi   = lane >= 16;

    const size_t headOff = ((size_t)b * NH + h) * NSEQ;
    const _Float16* qptr = Q + headOff * HD;
    const _Float16* kptr = K + headOff * HD;
    const _Float16* vptr = V + headOff * HD;
    const float*    bptr = bias + headOff * NSEQ;
    const unsigned char* mptr = mask + (size_t)b * NSEQ;

    const int q0 = qb * 128 + wid * 16;

    // staging assignments
    const int krow  = tid >> 2;          // 0..63
    const int kpart = (tid & 3) * 16;    // 0,16,32,48
    const int vd    = tid & 63;          // 0..63 (d column)
    const int vmq   = (tid >> 6) * 16;   // 0,16,32,48 (key range)

    // Q A-fragments (k = d: chunks 0..31, 32..63)
    v16h aq[2];
    {
        const _Float16* qr = qptr + (size_t)(q0 + lhalf) * HD;
#pragma unroll
        for (int c = 0; c < 2; ++c) {
            int db = c * 32 + (hi ? 8 : 0);
            aq[c] = mkv16(*(const h8*)(qr + db), *(const h8*)(qr + db + 16));
        }
    }

    v8f o[4];
#pragma unroll
    for (int i = 0; i < 4; ++i) o[i] = vzero8();
    float mrow[8], lrow[8];
#pragma unroll
    for (int r = 0; r < 8; ++r) { mrow[r] = -__builtin_inff(); lrow[r] = 0.0f; }

    for (int kb = 0; kb < 8; ++kb) {
        const int m0 = kb * 64;
        // prefetch next bias block: 32 lanes x 128B cover the 16x64 f32 tile
        if (kb < 7) {
            const float* pf = bptr + (size_t)(q0 + lhalf) * NSEQ + (m0 + 64) + (hi ? 32 : 0);
            __builtin_prefetch(pf, 0, 0);
        }
        __syncthreads();
        {   // K tile 64x64: contiguous 16-half chunks
            const h8* src = (const h8*)(kptr + (size_t)(m0 + krow) * HD + kpart);
#ifdef USE_ASYNC_COPY
            async_copy_b128(src,     &ldsK[krow][kpart]);
            async_copy_b128(src + 1, &ldsK[krow][kpart + 8]);
#else
            *(h8*)&ldsK[krow][kpart]     = src[0];
            *(h8*)&ldsK[krow][kpart + 8] = src[1];
#endif
        }
        {   // V tile 64x64 staged transposed: coalesced reads, vector stores
            h8 lo, hi8;
#pragma unroll
            for (int j = 0; j < 8; ++j)
                lo[j]  = vptr[(size_t)(m0 + vmq + j) * HD + vd];
#pragma unroll
            for (int j = 0; j < 8; ++j)
                hi8[j] = vptr[(size_t)(m0 + vmq + 8 + j) * HD + vd];
            *(h8*)&ldsVT[vd][vmq]     = lo;
            *(h8*)&ldsVT[vd][vmq + 8] = hi8;
        }
#ifdef USE_ASYNC_COPY
        __builtin_amdgcn_s_wait_asynccnt(0);
#endif
        __syncthreads();

        // S = Q . K^T  (16 x 64 per wave)
        v8f s[4];
#pragma unroll
        for (int nt = 0; nt < 4; ++nt) {
            v8f accS = vzero8();
            int mrel = nt * 16 + lhalf;
#pragma unroll
            for (int c = 0; c < 2; ++c) {
                int db = c * 32 + (hi ? 8 : 0);
                v16h bf = mkv16(*(const h8*)&ldsK[mrel][db], *(const h8*)&ldsK[mrel][db + 16]);
                accS = wmma_f16(aq[c], bf, accS);
            }
            s[nt] = accS;
        }

        // scale + bias + key padding mask
#pragma unroll
        for (int nt = 0; nt < 4; ++nt) {
            int mcol = m0 + nt * 16 + lhalf;
            unsigned char km = mptr[mcol];
#pragma unroll
            for (int r = 0; r < 8; ++r) {
                int qrow = q0 + r + (hi ? 8 : 0);
                float bv = bptr[(size_t)qrow * NSEQ + mcol];
                float val = s[nt][r] * 0.125f + bv;
                s[nt][r] = km ? -__builtin_inff() : val;
            }
        }

        // online softmax
        float mnew[8];
#pragma unroll
        for (int r = 0; r < 8; ++r)
            mnew[r] = fmaxf(fmaxf(s[0][r], s[1][r]), fmaxf(s[2][r], s[3][r]));
#pragma unroll
        for (int off = 8; off >= 1; off >>= 1)
#pragma unroll
            for (int r = 0; r < 8; ++r)
                mnew[r] = fmaxf(mnew[r], __shfl_xor(mnew[r], off, 32));

        float alpha[8];
#pragma unroll
        for (int r = 0; r < 8; ++r) {
            float mn = fmaxf(mrow[r], mnew[r]);
            alpha[r] = (mrow[r] == mn) ? 1.0f : __expf(mrow[r] - mn);
            mrow[r] = mn;
        }

        float psum[8];
#pragma unroll
        for (int r = 0; r < 8; ++r) psum[r] = 0.0f;
#pragma unroll
        for (int nt = 0; nt < 4; ++nt)
#pragma unroll
            for (int r = 0; r < 8; ++r) {
                float p = __expf(s[nt][r] - mrow[r]);
                s[nt][r] = p;
                psum[r] += p;
            }
#pragma unroll
        for (int off = 8; off >= 1; off >>= 1)
#pragma unroll
            for (int r = 0; r < 8; ++r)
                psum[r] += __shfl_xor(psum[r], off, 32);
#pragma unroll
        for (int r = 0; r < 8; ++r) lrow[r] = lrow[r] * alpha[r] + psum[r];
#pragma unroll
        for (int dt = 0; dt < 4; ++dt)
#pragma unroll
            for (int r = 0; r < 8; ++r) o[dt][r] *= alpha[r];

        // stage P (f16): C-frag layout -> A-frag layout via per-wave LDS
#pragma unroll
        for (int nt = 0; nt < 4; ++nt)
#pragma unroll
            for (int r = 0; r < 8; ++r)
                ldsP[wid][r + (hi ? 8 : 0)][nt * 16 + lhalf] = (_Float16)s[nt][r];
        __syncthreads();

        v16h ap[2];
#pragma unroll
        for (int mc = 0; mc < 2; ++mc) {
            int kb2 = mc * 32 + (hi ? 8 : 0);
            ap[mc] = mkv16(*(const h8*)&ldsP[wid][lhalf][kb2],
                           *(const h8*)&ldsP[wid][lhalf][kb2 + 16]);
        }
        // O += P . V
#pragma unroll
        for (int dt = 0; dt < 4; ++dt) {
            v8f accO = o[dt];
            int d = dt * 16 + lhalf;
#pragma unroll
            for (int mc = 0; mc < 2; ++mc) {
                int mb2 = mc * 32 + (hi ? 8 : 0);
                v16h bf = mkv16(*(const h8*)&ldsVT[d][mb2], *(const h8*)&ldsVT[d][mb2 + 16]);
                accO = wmma_f16(ap[mc], bf, accO);
            }
            o[dt] = accO;
        }
    }

    // normalize, write context f16: row = n*16 + b, col = h*64 + d
#pragma unroll
    for (int r = 0; r < 8; ++r) {
        float inv = 1.0f / lrow[r];
        int qrow = q0 + r + (hi ? 8 : 0);
        size_t rowI = (size_t)qrow * BATCH + b;
#pragma unroll
        for (int dt = 0; dt < 4; ++dt) {
            int d = dt * 16 + lhalf;
            ctx[rowI * HDIM + h * HD + d] = (_Float16)(o[dt][r] * inv);
        }
    }
}

// ---------------------------------------------------------------------------
// Kernel 3: output projection.  out = ctx(f16) @ Wo + bo  -> f32 (N,B,H)
// ---------------------------------------------------------------------------
__global__ __launch_bounds__(256) void oproj_kernel(
    const _Float16* __restrict__ ctx,
    const float* __restrict__ Wo, const float* __restrict__ bo,
    float* __restrict__ out)
{
    __shared__ alignas(16) _Float16 ldsX[64][40];
    __shared__ alignas(16) _Float16 ldsWT[128][40];

    const int mb = blockIdx.x;           // 0..127
    const int nb = blockIdx.y;           // 0..5
    const int ncc = nb * 128;

    const int tid   = threadIdx.x;
    const int wid   = tid >> 5;
    const int lane  = tid & 31;
    const int lhalf = lane & 15;
    const bool hi   = lane >= 16;
    const int wm    = wid & 1;
    const int wn    = wid >> 1;
    const int rowBase = mb * 64;

    const int xrow = tid >> 2;           // 0..63
    const int xc8  = (tid & 3) * 8;      // 0,8,16,24
    const int wcol = tid & 127;
    const int wkq  = (tid >> 7) * 16;

    v8f acc[2][2];
#pragma unroll
    for (int i = 0; i < 2; ++i)
#pragma unroll
        for (int j = 0; j < 2; ++j) acc[i][j] = vzero8();

    for (int k0 = 0; k0 < HDIM; k0 += 32) {
        __syncthreads();
        // ctx tile 64x32 f16: one h8 copy per thread
        *(h8*)&ldsX[xrow][xc8] =
            *(const h8*)(ctx + (size_t)(rowBase + xrow) * HDIM + k0 + xc8);
        // Wo tile 32x128 f32 -> f16, transposed
        {
            h8 lo, hi8;
#pragma unroll
            for (int j = 0; j < 8; ++j)
                lo[j]  = (_Float16)Wo[(size_t)(k0 + wkq + j) * HDIM + ncc + wcol];
#pragma unroll
            for (int j = 0; j < 8; ++j)
                hi8[j] = (_Float16)Wo[(size_t)(k0 + wkq + 8 + j) * HDIM + ncc + wcol];
            *(h8*)&ldsWT[wcol][wkq]     = lo;
            *(h8*)&ldsWT[wcol][wkq + 8] = hi8;
        }
        __syncthreads();

        v16h a[2];
#pragma unroll
        for (int fm = 0; fm < 2; ++fm) {
            int row = wm * 32 + fm * 16 + lhalf;
            int db  = hi ? 8 : 0;
            a[fm] = mkv16(*(const h8*)&ldsX[row][db], *(const h8*)&ldsX[row][db + 16]);
        }
        v16h bf[2];
#pragma unroll
        for (int fn = 0; fn < 2; ++fn) {
            int col = wn * 32 + fn * 16 + lhalf;
            int kb2 = hi ? 8 : 0;
            bf[fn] = mkv16(*(const h8*)&ldsWT[col][kb2], *(const h8*)&ldsWT[col][kb2 + 16]);
        }
#pragma unroll
        for (int fm = 0; fm < 2; ++fm)
#pragma unroll
            for (int fn = 0; fn < 2; ++fn)
                acc[fm][fn] = wmma_f16(a[fm], bf[fn], acc[fm][fn]);
    }

#pragma unroll
    for (int fm = 0; fm < 2; ++fm) {
#pragma unroll
        for (int fn = 0; fn < 2; ++fn) {
            int col = ncc + wn * 32 + fn * 16 + lhalf;
            float bb = bo[col];
#pragma unroll
            for (int r = 0; r < 8; ++r) {
                int rowI = rowBase + wm * 32 + fm * 16 + r + (hi ? 8 : 0);
                out[(size_t)rowI * HDIM + col] = acc[fm][fn][r] + bb;
            }
        }
    }
}

// ---------------------------------------------------------------------------
extern "C" void kernel_launch(void* const* d_in, const int* in_sizes, int n_in,
                              void* d_out, int out_size, void* d_ws, size_t ws_size,
                              hipStream_t stream) {
    const float* x    = (const float*)d_in[0];
    const float* abias= (const float*)d_in[1];
    const unsigned char* mask = (const unsigned char*)d_in[2];
    const float* Wq = (const float*)d_in[3];
    const float* bq = (const float*)d_in[4];
    const float* Wk = (const float*)d_in[5];
    const float* bk = (const float*)d_in[6];
    const float* Wv = (const float*)d_in[7];
    const float* bv = (const float*)d_in[8];
    const float* Wo = (const float*)d_in[9];
    const float* bo = (const float*)d_in[10];
    float* out = (float*)d_out;

    const size_t headElems = (size_t)BATCH * NH * NSEQ * HD;  // 6,291,456
    _Float16* Qb  = (_Float16*)d_ws;
    _Float16* Kb  = Qb + headElems;
    _Float16* Vb  = Kb + headElems;
    _Float16* Cb  = Vb + headElems;       // ctx: 8192*768 halves

    qkv_kernel<<<dim3(128, 18), 256, 0, stream>>>(x, Wq, Wk, Wv, bq, bk, bv, Qb, Kb, Vb);
    attn_kernel<<<dim3(768), 256, 0, stream>>>(Qb, Kb, Vb, abias, mask, Cb);
    oproj_kernel<<<dim3(128, 6), 256, 0, stream>>>(Cb, Wo, bo, out);
}
